// SpikingLayer_9594956939812
// MI455X (gfx1250) — matrix operand order repaired and checked by reading, not compiled, not run on hardware
//
#include <hip/hip_runtime.h>
#include <hip/hip_bf16.h>
#include <stddef.h>

#define NN 8192
#define STEPS 64
#define DECAYF 0.9f
#define THRESHF 1.0f

#define WAVES_PER_BLOCK 4
#define ROWS_PER_WAVE 16
#define ROWS_PER_BLOCK (WAVES_PER_BLOCK * ROWS_PER_WAVE)   // 64
#define ROW_BLOCKS (NN / ROWS_PER_BLOCK)                   // 128
#define KSPLIT 4
#define KSEG (NN / KSPLIT)                                 // 2048

typedef __attribute__((ext_vector_type(16))) __bf16       v16bf;
typedef __attribute__((ext_vector_type(8)))  float        v8f;
typedef __attribute__((ext_vector_type(4)))  unsigned int v4u;

union AFrag { v16bf v; v4u u[2]; __bf16 h[16]; };
union Pack8 { __bf16 h[8]; v4u u; };
union F4    { v4u u; float f[4]; };

// ---------------------------------------------------------------------------
// Per-call state init: f double-buffer slot 0, membrane potential v, u := 0.
// (harness poisons d_ws, so every kernel_launch call must rebuild all state)
// ---------------------------------------------------------------------------
__global__ void init_state(const float* __restrict__ f0, const float* __restrict__ v0,
                           __bf16* __restrict__ fb, float* __restrict__ v,
                           float* __restrict__ u) {
  int i = blockIdx.x * blockDim.x + threadIdx.x;
  if (i < NN) {
    fb[i] = (__bf16)f0[i];
    v[i]  = v0[i];
    u[i]  = 0.0f;
  }
}

// ---------------------------------------------------------------------------
// One-time (per call) f32 -> bf16 conversion of W. NT loads on the f32 stream
// so the 256 MB source pass does not evict the 128 MB bf16 image we want to
// keep resident in the 192 MB L2 for all 64 matvec passes.
// ---------------------------------------------------------------------------
__global__ void convert_w(const float* __restrict__ W, __bf16* __restrict__ Wb) {
  const size_t nchunks = (size_t)NN * (size_t)NN / 8;
  const size_t stride  = (size_t)gridDim.x * blockDim.x;
  for (size_t i = (size_t)blockIdx.x * blockDim.x + threadIdx.x; i < nchunks; i += stride) {
    F4 a, b;
    a.u = __builtin_nontemporal_load(reinterpret_cast<const v4u*>(W) + 2 * i);
    b.u = __builtin_nontemporal_load(reinterpret_cast<const v4u*>(W) + 2 * i + 1);
    Pack8 p;
    p.h[0] = (__bf16)a.f[0]; p.h[1] = (__bf16)a.f[1];
    p.h[2] = (__bf16)a.f[2]; p.h[3] = (__bf16)a.f[3];
    p.h[4] = (__bf16)b.f[0]; p.h[5] = (__bf16)b.f[1];
    p.h[6] = (__bf16)b.f[2]; p.h[7] = (__bf16)b.f[3];
    reinterpret_cast<v4u*>(Wb)[i] = p.u;
  }
}

// ---------------------------------------------------------------------------
// Partial matvec u += W[rows, k0:k1] @ f[k0:k1] via v_wmma_f32_16x16x32_bf16.
// B has f broadcast into all 16 columns => every D column equals u[m]; lane 0
// holds u[row_base+0..7] in acc[0..7], lane 16 holds u[row_base+8..15].
// A fragment (16-bit A 16x32 layout):
//   lanes 0-15  (M=lane):    V0-3 = K 0..7,  V4-7 = K 16..23
//   lanes 16-31 (M=lane-16): V0-3 = K 8..15, V4-7 = K 24..31
// -> two 16B b128 loads per lane from the row-major bf16 W row.
// B fragment: same K layout as A -> two 16B loads (K 0..7,16..23 / 8..15,24..31).
// ---------------------------------------------------------------------------
template <bool WB16>
__global__ __launch_bounds__(WAVES_PER_BLOCK * 32)
void matvec_partial(const void* __restrict__ Wsrc, const __bf16* __restrict__ f_in,
                    float* __restrict__ u) {
  const int lane = threadIdx.x & 31;
  const int wave = threadIdx.x >> 5;
  const int rb   = blockIdx.x / KSPLIT;
  const int ks   = blockIdx.x % KSPLIT;
  const int row_base = (rb * WAVES_PER_BLOCK + wave) * ROWS_PER_WAVE;
  const int m    = lane & 15;
  const int hi   = lane >> 4;    // 0: lanes 0-15, 1: lanes 16-31
  const int a_kh = hi * 8;       // A K-half offset (elements)
  const int b_ko = hi * 8;       // B K-half offset (elements)
  const int k0   = ks * KSEG;
  const int k1   = k0 + KSEG;

  v8f acc = {};

  const __bf16* bp = f_in + b_ko;

  if constexpr (WB16) {
    const __bf16* ap = (const __bf16*)Wsrc + (size_t)(row_base + m) * NN + a_kh;
#pragma unroll 2
    for (int k = k0; k < k1; k += 32) {
      AFrag a, b;
      a.u[0] = *reinterpret_cast<const v4u*>(ap + k);
      a.u[1] = *reinterpret_cast<const v4u*>(ap + k + 16);
      b.u[0] = *reinterpret_cast<const v4u*>(bp + k);
      b.u[1] = *reinterpret_cast<const v4u*>(bp + k + 16);
#if __HIP_DEVICE_COMPILE__
      acc = __builtin_amdgcn_wmma_f32_16x16x32_bf16(
          false, a.v, false, b.v, (short)0, acc, false, false);
#endif
    }
  } else {
    // Fallback (ws too small for bf16 W): stream f32 W, convert in-register.
    const float* ap = (const float*)Wsrc + (size_t)(row_base + m) * NN + a_kh;
#pragma unroll 2
    for (int k = k0; k < k1; k += 32) {
      float4 x0 = *reinterpret_cast<const float4*>(ap + k);
      float4 x1 = *reinterpret_cast<const float4*>(ap + k + 4);
      float4 x2 = *reinterpret_cast<const float4*>(ap + k + 16);
      float4 x3 = *reinterpret_cast<const float4*>(ap + k + 20);
      AFrag a, b;
      a.h[0]  = (__bf16)x0.x; a.h[1]  = (__bf16)x0.y;
      a.h[2]  = (__bf16)x0.z; a.h[3]  = (__bf16)x0.w;
      a.h[4]  = (__bf16)x1.x; a.h[5]  = (__bf16)x1.y;
      a.h[6]  = (__bf16)x1.z; a.h[7]  = (__bf16)x1.w;
      a.h[8]  = (__bf16)x2.x; a.h[9]  = (__bf16)x2.y;
      a.h[10] = (__bf16)x2.z; a.h[11] = (__bf16)x2.w;
      a.h[12] = (__bf16)x3.x; a.h[13] = (__bf16)x3.y;
      a.h[14] = (__bf16)x3.z; a.h[15] = (__bf16)x3.w;
      b.u[0] = *reinterpret_cast<const v4u*>(bp + k);
      b.u[1] = *reinterpret_cast<const v4u*>(bp + k + 16);
#if __HIP_DEVICE_COMPILE__
      acc = __builtin_amdgcn_wmma_f32_16x16x32_bf16(
          false, a.v, false, b.v, (short)0, acc, false, false);
#endif
    }
  }

  // D layout: VGPR r, lanes 0-15 -> M=r, lanes 16-31 -> M=8+r. All columns
  // identical, so lane 0 / lane 16 carry the 16 row results.
  if (m == 0) {
    const int r0 = row_base + hi * 8;
#pragma unroll
    for (int r = 0; r < 8; ++r)
      atomicAdd(&u[r0 + r], acc[r]);
  }
}

// ---------------------------------------------------------------------------
// LIF update: v = 0.9 v + u; spike = v >= 1; reset on spike; emit f (bf16 for
// the next matvec, f32 into the firing matrix); re-zero u for the next step.
// ---------------------------------------------------------------------------
__global__ void lif_update(float* __restrict__ u, float* __restrict__ v,
                           __bf16* __restrict__ f_out, float* __restrict__ out_row) {
  int i = blockIdx.x * blockDim.x + threadIdx.x;
  if (i < NN) {
    float vv    = DECAYF * v[i] + u[i];
    bool  spike = vv >= THRESHF;
    float fn    = spike ? 1.0f : 0.0f;
    v[i]      = spike ? 0.0f : vv;
    f_out[i]  = (__bf16)fn;
    out_row[i] = fn;
    u[i] = 0.0f;
  }
}

extern "C" void kernel_launch(void* const* d_in, const int* in_sizes, int n_in,
                              void* d_out, int out_size, void* d_ws, size_t ws_size,
                              hipStream_t stream) {
  (void)in_sizes; (void)n_in; (void)out_size;

  const float* W  = (const float*)d_in[0];
  const float* f0 = (const float*)d_in[1];
  const float* v0 = (const float*)d_in[2];
  float* out = (float*)d_out;

  // Workspace layout: [f double-buffer bf16 2*NN][v f32 NN][u f32 NN][Wb bf16 NN*NN]
  char*   ws = (char*)d_ws;
  __bf16* fb = (__bf16*)ws;
  float*  v  = (float*)(ws + (size_t)2 * NN * sizeof(__bf16));
  float*  u  = v + NN;
  size_t  off = (size_t)((char*)(u + NN) - ws);
  off = (off + 255) & ~(size_t)255;
  __bf16* Wb = (__bf16*)(ws + off);
  const bool use_bf16 = ws_size >= off + (size_t)NN * (size_t)NN * sizeof(__bf16);

  init_state<<<(NN + 255) / 256, 256, 0, stream>>>(f0, v0, fb, v, u);
  if (use_bf16)
    convert_w<<<4096, 256, 0, stream>>>(W, Wb);

  for (int t = 0; t < STEPS; ++t) {
    const __bf16* fin  = fb + (size_t)(t & 1) * NN;
    __bf16*       fout = fb + (size_t)((t + 1) & 1) * NN;
    if (use_bf16)
      matvec_partial<true><<<ROW_BLOCKS * KSPLIT, WAVES_PER_BLOCK * 32, 0, stream>>>(
          (const void*)Wb, fin, u);
    else
      matvec_partial<false><<<ROW_BLOCKS * KSPLIT, WAVES_PER_BLOCK * 32, 0, stream>>>(
          (const void*)W, fin, u);
    lif_update<<<(NN + 255) / 256, 256, 0, stream>>>(u, v, fout, out + (size_t)t * NN);
  }
}
